// GCN_42795054137778
// MI455X (gfx1250) — matrix operand order repaired and checked by reading, not compiled, not run on hardware
//
#include <hip/hip_runtime.h>
#include <hip/hip_bf16.h>

#define N_NODES 100000
#define N_EDGES 1600000
#define HID     128
#define NGRAPH  1000
#define NCLS    10
#define NLAYER  3
#define BN_EPS  1e-5f

typedef __attribute__((ext_vector_type(16))) __bf16 v16bf;
typedef __attribute__((ext_vector_type(8)))  __bf16 v8bf;
typedef __attribute__((ext_vector_type(8)))  float  v8f;
typedef __attribute__((ext_vector_type(4)))  int    v4i;

#if defined(__has_builtin)
#if __has_builtin(__builtin_amdgcn_global_load_async_to_lds_b128) && \
    __has_builtin(__builtin_amdgcn_s_wait_asynccnt)
#define HAVE_ASYNC_LDS 1
#endif
#endif

#ifdef HAVE_ASYNC_LDS
typedef __attribute__((address_space(1))) v4i g_v4i;  // global (HIP "__device__")
typedef __attribute__((address_space(3))) v4i l_v4i;  // LDS
#endif

// ---------------- generic zero ----------------
__global__ void k_zero(float* p, int n) {
  int i = blockIdx.x * blockDim.x + threadIdx.x;
  if (i < n) p[i] = 0.f;
}

// ---------------- degree / dinv ----------------
__global__ void k_degree(const long long* __restrict__ ei, float* cnt) {
  int e = blockIdx.x * blockDim.x + threadIdx.x;
  if (e < N_EDGES) {
    int d = (int)ei[(size_t)N_EDGES + e];
    atomicAdd(&cnt[d], 1.0f);
  }
}
__global__ void k_dinv(float* dinv) {
  int i = blockIdx.x * blockDim.x + threadIdx.x;
  if (i < N_NODES) dinv[i] = rsqrtf(1.0f + dinv[i]);
}

// ---- convert + transpose all layer weights to bf16 once: Wt[l][n*128+k] = W[l][k][n] ----
__global__ void k_convert_wT(const float* __restrict__ W, __bf16* __restrict__ Wt) {
  int i = blockIdx.x * blockDim.x + threadIdx.x;
  if (i >= NLAYER * HID * HID) return;
  int l = i >> 14;
  int r = i & 16383;
  int n = r >> 7, k = r & 127;
  Wt[i] = (__bf16)W[(size_t)l * HID * HID + (size_t)k * HID + n];
}

// ---------------- WMMA GEMM: Out[N,128] = A[N,128] @ W[128,128] ----------------
// 256 threads = 8 waves/block; each wave owns a 16-row tile x all 128 cols.
// Pre-transposed bf16 W tile (32 KB) is pulled into LDS, via the gfx1250
// async-to-LDS path (ASYNCcnt) when available.
__global__ __launch_bounds__(256) void k_gemm_wmma(const float* __restrict__ A,
                                                   const __bf16* __restrict__ Wt,
                                                   float* __restrict__ Out,
                                                   int nRowTiles) {
  __shared__ __attribute__((aligned(16))) __bf16 wT[HID * HID];  // col-major W
#ifdef HAVE_ASYNC_LDS
  for (int i = threadIdx.x * 8; i < HID * HID; i += 256 * 8) {
    __builtin_amdgcn_global_load_async_to_lds_b128(
        (g_v4i*)(Wt + i), (l_v4i*)(&wT[i]), 0, 0);
  }
  __builtin_amdgcn_s_wait_asynccnt(0);
  __syncthreads();
#else
  for (int i = threadIdx.x * 8; i < HID * HID; i += 256 * 8) {
    *(v8bf*)&wT[i] = *(const v8bf*)(Wt + i);
  }
  __syncthreads();
#endif

  int wave = threadIdx.x >> 5;
  int lane = threadIdx.x & 31;
  int rowTile = blockIdx.x * 8 + wave;
  if (rowTile >= nRowTiles) return;  // wave-uniform; EXEC all-1 around WMMA

  int m = lane & 15;
  const float* arow = A + (size_t)(rowTile * 16 + m) * HID;
  int koffA = (lane < 16) ? 0 : 16;   // contiguous k order, matches B
  int nB    = lane & 15;
  int koffB = (lane < 16) ? 0 : 16;   // contiguous k order, same as A

  v8f acc[8] = {};
  for (int kk = 0; kk < HID; kk += 32) {
    v16bf afrag;
#pragma unroll
    for (int e2 = 0; e2 < 8; ++e2) {
      afrag[e2]     = (__bf16)arow[kk + koffA + e2];        // K = koff..koff+7
      afrag[e2 + 8] = (__bf16)arow[kk + 8 + koffA + e2];    // K = 8+koff..koff+15
    }
#pragma unroll
    for (int ct = 0; ct < 8; ++ct) {
      union { v16bf v; v8bf h[2]; } b;
      const v8bf* bp = (const v8bf*)&wT[(ct * 16 + nB) * HID + kk + koffB];
      b.h[0] = bp[0];
      b.h[1] = bp[1];
      acc[ct] = __builtin_amdgcn_wmma_f32_16x16x32_bf16(
          false, afrag, false, b.v, (short)0, acc[ct], false, false);
    }
  }
  // C/D layout: VGPR r -> row (lane<16 ? r : 8+r), col = lane&15
  int rowBase = rowTile * 16 + ((lane < 16) ? 0 : 8);
#pragma unroll
  for (int ct = 0; ct < 8; ++ct) {
#pragma unroll
    for (int r = 0; r < 8; ++r) {
      Out[(size_t)(rowBase + r) * HID + ct * 16 + nB] = acc[ct][r];
    }
  }
}

// ---------------- edge aggregation: one wave per edge, float4 per lane ----------------
__global__ void k_edge_agg(const long long* __restrict__ ei,
                           const float* __restrict__ dinv,
                           const float* __restrict__ h,
                           float* agg) {
  int gid = blockIdx.x * blockDim.x + threadIdx.x;
  int e = gid >> 5;
  if (e >= N_EDGES) return;
  int lane = gid & 31;
  int s = (int)ei[e];
  int d = (int)ei[(size_t)N_EDGES + e];
  float nrm = dinv[s] * dinv[d];
  const float4 hv = *(const float4*)(h + (size_t)s * HID + lane * 4);
  float* ap = agg + (size_t)d * HID + lane * 4;
  atomicAdd(ap + 0, hv.x * nrm);
  atomicAdd(ap + 1, hv.y * nrm);
  atomicAdd(ap + 2, hv.z * nrm);
  atomicAdd(ap + 3, hv.w * nrm);
}

// ---------------- self-loop + bias ----------------
__global__ void k_selfloop_bias(const float* __restrict__ h,
                                const float* __restrict__ dinv,
                                const float* __restrict__ b, float* agg) {
  int i = blockIdx.x * blockDim.x + threadIdx.x;
  if (i >= N_NODES * HID) return;
  int n = i >> 7, f = i & 127;
  float di = dinv[n];
  agg[i] += h[i] * di * di + b[f];
}

// ---------------- BN: column sums/sumsq via LDS atomics ----------------
__global__ void k_bn_stats(const float* __restrict__ h, float* stats) {
  __shared__ float ss[2 * HID];
  ss[threadIdx.x] = 0.f;  // blockDim == 256 == 2*HID
  __syncthreads();
  int total = N_NODES * HID;
  for (int i = blockIdx.x * blockDim.x + threadIdx.x; i < total;
       i += gridDim.x * blockDim.x) {
    float v = h[i];
    int f = i & 127;
    atomicAdd(&ss[f], v);
    atomicAdd(&ss[HID + f], v * v);
  }
  __syncthreads();
  atomicAdd(&stats[threadIdx.x], ss[threadIdx.x]);
}
__global__ void k_bn_finalize(const float* __restrict__ stats,
                              const float* __restrict__ gamma,
                              const float* __restrict__ beta, float* scsh) {
  int f = threadIdx.x;
  if (f < HID) {
    float mean = stats[f] * (1.0f / N_NODES);
    float var  = fmaxf(stats[HID + f] * (1.0f / N_NODES) - mean * mean, 0.f);
    float sc   = gamma[f] * rsqrtf(var + BN_EPS);
    scsh[f] = sc;
    scsh[HID + f] = beta[f] - mean * sc;
  }
}
__global__ void k_bn_relu(float* h, const float* __restrict__ scsh) {
  int i = blockIdx.x * blockDim.x + threadIdx.x;
  if (i >= N_NODES * HID) return;
  int f = i & 127;
  float v = fmaf(h[i], scsh[f], scsh[HID + f]);
  h[i] = v > 0.f ? v : 0.f;
}

// ---------------- pooling ----------------
__global__ void k_pool_feat(const float* __restrict__ h,
                            const long long* __restrict__ batch, float* pooled) {
  int i = blockIdx.x * blockDim.x + threadIdx.x;
  if (i >= N_NODES * HID) return;
  int n = i >> 7;
  int g = (int)batch[n];
  atomicAdd(&pooled[(size_t)g * HID + (i & 127)], h[i]);
}
__global__ void k_pool_cnt(const long long* __restrict__ batch, float* cnt) {
  int n = blockIdx.x * blockDim.x + threadIdx.x;
  if (n < N_NODES) atomicAdd(&cnt[(int)batch[n]], 1.0f);
}

// ---------------- classifier MLP ----------------
__global__ void k_mlp1(const float* __restrict__ pooled,
                       const float* __restrict__ cnt,
                       const float* __restrict__ w1,
                       const float* __restrict__ b1, float* z) {
  int i = blockIdx.x * blockDim.x + threadIdx.x;
  if (i >= NGRAPH * 64) return;
  int g = i >> 6, j = i & 63;
  float inv = 1.0f / fmaxf(cnt[g], 1.0f);
  const float* pg = pooled + (size_t)g * HID;
  float s = 0.f;
  for (int k = 0; k < HID; ++k) s = fmaf(pg[k], w1[k * 64 + j], s);
  float v = s * inv + b1[j];
  z[i] = v > 0.f ? v : 0.f;
}
__global__ void k_mlp2(const float* __restrict__ z,
                       const float* __restrict__ w2,
                       const float* __restrict__ b2, float* out) {
  int i = blockIdx.x * blockDim.x + threadIdx.x;
  if (i >= NGRAPH * NCLS) return;
  int g = i / NCLS, c = i % NCLS;
  const float* zg = z + (size_t)g * 64;
  float s = b2[c];
  for (int j = 0; j < 64; ++j) s = fmaf(zg[j], w2[j * NCLS + c], s);
  out[i] = s;
}

extern "C" void kernel_launch(void* const* d_in, const int* in_sizes, int n_in,
                              void* d_out, int out_size, void* d_ws, size_t ws_size,
                              hipStream_t stream) {
  (void)in_sizes; (void)n_in; (void)out_size; (void)ws_size;
  const float*     x      = (const float*)d_in[0];
  const float*     conv_w = (const float*)d_in[1];
  const float*     conv_b = (const float*)d_in[2];
  const float*     bn_g   = (const float*)d_in[3];
  const float*     bn_b   = (const float*)d_in[4];
  const float*     w1     = (const float*)d_in[5];
  const float*     b1     = (const float*)d_in[6];
  const float*     w2     = (const float*)d_in[7];
  const float*     b2     = (const float*)d_in[8];
  const long long* ei     = (const long long*)d_in[9];   // int64 edge_index [2,E]
  const long long* batch  = (const long long*)d_in[10];  // int64 [N]
  float* out = (float*)d_out;

  char* wsp = (char*)d_ws; size_t off = 0;
  auto take = [&](size_t bytes) {
    void* p = wsp + off;
    off = (off + bytes + 255) & ~(size_t)255;
    return p;
  };
  float*  hw     = (float*)take((size_t)N_NODES * HID * 4);  // h = x @ W
  float*  agg    = (float*)take((size_t)N_NODES * HID * 4);  // aggregated / activations
  float*  dinv   = (float*)take((size_t)N_NODES * 4);
  float*  stats  = (float*)take(2 * HID * 4);
  float*  scsh   = (float*)take(2 * HID * 4);
  float*  pooled = (float*)take((size_t)NGRAPH * HID * 4);
  float*  cnt    = (float*)take((size_t)NGRAPH * 4);
  float*  zbuf   = (float*)take((size_t)NGRAPH * 64 * 4);
  __bf16* wTg    = (__bf16*)take((size_t)NLAYER * HID * HID * 2);

  const int T = 256;
  auto cdiv = [](long long a, long long b) { return (int)((a + b - 1) / b); };
  const int NH = N_NODES * HID;

  // one-time weight convert+transpose (bf16, col-major)
  k_convert_wT<<<cdiv(NLAYER * HID * HID, T), T, 0, stream>>>(conv_w, wTg);

  // degrees -> dinv = rsqrt(1 + deg)
  k_zero<<<cdiv(N_NODES, T), T, 0, stream>>>(dinv, N_NODES);
  k_degree<<<cdiv(N_EDGES, T), T, 0, stream>>>(ei, dinv);
  k_dinv<<<cdiv(N_NODES, T), T, 0, stream>>>(dinv);

  const int nRowTiles = N_NODES / 16;  // 6250, exact
  const float* cur = x;
  for (int l = 0; l < NLAYER; ++l) {
    k_gemm_wmma<<<cdiv(nRowTiles, 8), T, 0, stream>>>(
        cur, wTg + (size_t)l * HID * HID, hw, nRowTiles);
    k_zero<<<cdiv(NH, T), T, 0, stream>>>(agg, NH);
    k_edge_agg<<<cdiv((long long)N_EDGES * 32, T), T, 0, stream>>>(ei, dinv, hw, agg);
    k_selfloop_bias<<<cdiv(NH, T), T, 0, stream>>>(hw, dinv, conv_b + l * HID, agg);
    k_zero<<<1, 2 * HID, 0, stream>>>(stats, 2 * HID);
    k_bn_stats<<<1024, T, 0, stream>>>(agg, stats);
    k_bn_finalize<<<1, HID, 0, stream>>>(stats, bn_g + l * HID, bn_b + l * HID, scsh);
    k_bn_relu<<<cdiv(NH, T), T, 0, stream>>>(agg, scsh);
    cur = agg;
  }

  // global mean pool + MLP
  k_zero<<<cdiv(NGRAPH * HID, T), T, 0, stream>>>(pooled, NGRAPH * HID);
  k_zero<<<cdiv(NGRAPH, T), T, 0, stream>>>(cnt, NGRAPH);
  k_pool_feat<<<cdiv(NH, T), T, 0, stream>>>(agg, batch, pooled);
  k_pool_cnt<<<cdiv(N_NODES, T), T, 0, stream>>>(batch, cnt);
  k_mlp1<<<cdiv(NGRAPH * 64, T), T, 0, stream>>>(pooled, cnt, w1, b1, zbuf);
  k_mlp2<<<cdiv(NGRAPH * NCLS, T), T, 0, stream>>>(zbuf, w2, b2, out);
}
